// MultiheadAttention_48576080118149
// MI455X (gfx1250) — matrix operand order repaired and checked by reading, not compiled, not run on hardware
//
#include <hip/hip_runtime.h>

// ---- problem constants -----------------------------------------------------
constexpr int Bb = 2, Ss = 2048, Ee = 1024, Hh = 16, Dd = 64;
constexpr int Mrows = Bb * Ss;              // 4096
constexpr float INV_SCALE = 0.125f;         // 1/sqrt(64), folded into Q proj

// ---- WMMA fragment types ---------------------------------------------------
typedef __attribute__((ext_vector_type(16))) __bf16 v16bf;
typedef __attribute__((ext_vector_type(8)))  float  v8f;
typedef __attribute__((ext_vector_type(4)))  int    v4i;
typedef __attribute__((ext_vector_type(8)))  int    v8i;

union FragU {
    v16bf v;
    uint4 q[2];
};

__device__ __forceinline__ unsigned short f2bfu(float f) {
    unsigned int u = __float_as_uint(f);
    u += 0x7FFFu + ((u >> 16) & 1u);            // round-to-nearest-even
    return (unsigned short)(u >> 16);
}

// packed pair convert: float,float -> 2 x bf16 in one dword
__device__ __forceinline__ unsigned int f2bf2(float lo, float hi) {
#if __has_builtin(__builtin_amdgcn_cvt_pk_bf16_f32)
    typedef __attribute__((ext_vector_type(2))) __bf16 v2bf;
    union { v2bf v; unsigned int u; } t;
    t.v = __builtin_amdgcn_cvt_pk_bf16_f32(lo, hi);
    return t.u;
#else
    return (unsigned int)f2bfu(lo) | ((unsigned int)f2bfu(hi) << 16);
#endif
}

__device__ __forceinline__ v8f wmma_bf16(const FragU& a, const FragU& b, v8f c) {
    return __builtin_amdgcn_wmma_f32_16x16x32_bf16(
        false, a.v, false, b.v, (short)0, c, false, false);
}

// ----------------------------------------------------------------------------
// Tensor Data Mover: 2D tile load global->LDS (CDNA5 ISA).
// D# group0: [1:0]=count=1, [63:32]=lds_addr, [120:64]=global_addr, [127:126]=2
// D# group1: [17:16]=data_size(1 => 2B), [79:48]=tensor_dim0, [111:80]=tensor_dim1,
//            [127:112]=tile_dim0, [143:128]=tile_dim1, [207:160]=dim0_stride
// Tile rows are packed contiguously into LDS (X fastest, then Y).
// ----------------------------------------------------------------------------
__device__ __forceinline__ void tdm_load_2d(unsigned lds_off,
                                            unsigned long long gaddr,
                                            int tile_x, int tile_y,
                                            int tensor_x, int tensor_y,
                                            int stride_x)
{
    v4i g0;
    g0[0] = 1;                                            // count=1 (valid)
    g0[1] = (int)lds_off;                                 // lds_addr (bytes)
    g0[2] = (int)(gaddr & 0xFFFFFFFFull);                 // global_addr[31:0]
    g0[3] = (int)(((gaddr >> 32) & 0x01FFFFFFull) | 0x80000000u); // [56:32]|type=2

    v8i g1;
    g1[0] = 1 << 16;                                      // data_size=2B, mask=0
    g1[1] = (tensor_x & 0xFFFF) << 16;                    // tensor_dim0[15:0]
    g1[2] = ((tensor_x >> 16) & 0xFFFF) | ((tensor_y & 0xFFFF) << 16);
    g1[3] = ((tensor_y >> 16) & 0xFFFF) | ((tile_x & 0xFFFF) << 16);
    g1[4] = tile_y & 0xFFFF;                              // tile_dim1, tile_dim2=0
    g1[5] = stride_x;                                     // dim0_stride[31:0]
    g1[6] = 0;
    g1[7] = 0;
    asm volatile("tensor_load_to_lds %0, %1" :: "s"(g0), "s"(g1) : "memory");
}

// ============================================================================
// fp32 -> bf16 bulk convert (8 elements / thread)
// ============================================================================
__global__ __launch_bounds__(256)
void to_bf16(const float* __restrict__ in, unsigned short* __restrict__ out,
             int n8)
{
    int i = blockIdx.x * 256 + threadIdx.x;
    if (i >= n8) return;
    const float4* p = (const float4*)in + (size_t)i * 2;
    float4 f0 = p[0], f1 = p[1];
    uint4 t;
    t.x = f2bf2(f0.x, f0.y); t.y = f2bf2(f0.z, f0.w);
    t.z = f2bf2(f1.x, f1.y); t.w = f2bf2(f1.z, f1.w);
    ((uint4*)out)[i] = t;
}

// ============================================================================
// Projection GEMM (all-bf16 inputs): Y[M,N] = s * (A[M,K] @ W[N,K]^T)
// M=4096, N=K=1024. Block tile 64(M) x 128(N), 256 threads = 8 waves (2x4),
// wave tile 32x32 = 2x2 WMMA accumulators. K-step 64.
// Tiles staged by the Tensor Data Mover, DOUBLE-BUFFERED: wave 0 issues the
// next tile's two TDM ops, then s_wait_tensorcnt(2) so the in-flight pair
// overlaps this tile's WMMAs (TDM ops retire in order per wave).
// Dynamic LDS (base offset 0), per buffer: As 64x64, Bs 128x64 (24 KB x 2).
// OUT_MODE: 0 = bf16 row-major, 1 = bf16 transposed-V [b,h,d,s], 2 = fp32
// ============================================================================
template<int OUT_MODE>
__global__ __launch_bounds__(256)
void proj_gemm(const unsigned short* __restrict__ A,
               const unsigned short* __restrict__ W,
               void* __restrict__ Out, float out_scale)
{
    extern __shared__ unsigned short smem[];
    constexpr unsigned BUF_ELEMS = 4096 + 8192;          // As + Bs per buffer
    constexpr unsigned LDS_A0 = 0,              LDS_B0 = 4096 * 2;
    constexpr unsigned LDS_A1 = BUF_ELEMS * 2,  LDS_B1 = BUF_ELEMS * 2 + 4096 * 2;

    const int tid  = threadIdx.x;
    const int lane = tid & 31;
    const int wave = tid >> 5;
    const int wm   = wave >> 2;              // 0..1
    const int wn   = wave & 3;               // 0..3
    const int ln   = lane & 15;
    const int lh   = lane >> 4;
    const int Mt   = blockIdx.y * 64;
    const int Nt   = blockIdx.x * 128;

    const unsigned long long Abase =
        (unsigned long long)(const void*)&A[(size_t)Mt * Ee];
    const unsigned long long Wbase =
        (unsigned long long)(const void*)&W[(size_t)Nt * Ee];

    v8f acc[2][2] = {};

    // ---- prologue: stage tile 0 into buffer 0 ----
    if (wave == 0) {
        tdm_load_2d(LDS_A0, Abase, 64, 64,  Ee, Mrows, Ee);
        tdm_load_2d(LDS_B0, Wbase, 64, 128, Ee, Ee,    Ee);
    }

    for (int k0 = 0; k0 < Ee; k0 += 64) {
        const int buf = (k0 >> 6) & 1;
        const unsigned short (*As)[64] =
            (const unsigned short(*)[64])(smem + buf * BUF_ELEMS);
        const unsigned short (*Bs)[64] =
            (const unsigned short(*)[64])(smem + buf * BUF_ELEMS + 4096);

        if (wave == 0) {
            if (k0 + 64 < Ee) {
                // issue next tile into the other buffer, then wait until only
                // those two TDM ops remain outstanding => current tile ready
                tdm_load_2d(buf ? LDS_A0 : LDS_A1, Abase + (size_t)(k0 + 64) * 2,
                            64, 64,  Ee, Mrows, Ee);
                tdm_load_2d(buf ? LDS_B0 : LDS_B1, Wbase + (size_t)(k0 + 64) * 2,
                            64, 128, Ee, Ee,    Ee);
                __builtin_amdgcn_s_wait_tensorcnt(2);
            } else {
                __builtin_amdgcn_s_wait_tensorcnt(0);
            }
        }
        __syncthreads();

        // ---- 2 x (fragment load + 2x2 WMMA) over the 64-wide K step ----
#pragma unroll
        for (int kk = 0; kk < 2; ++kk) {
            FragU af[2], bfr[2];
#pragma unroll
            for (int i = 0; i < 2; ++i) {
                int row = wm * 32 + i * 16 + ln;
                af[i].q[0] = *(const uint4*)&As[row][kk * 32 + lh * 8];
                af[i].q[1] = *(const uint4*)&As[row][kk * 32 + 16 + lh * 8];
            }
#pragma unroll
            for (int j = 0; j < 2; ++j) {
                int col = wn * 32 + j * 16 + ln;
                bfr[j].q[0] = *(const uint4*)&Bs[col][kk * 32 + lh * 8];
                bfr[j].q[1] = *(const uint4*)&Bs[col][kk * 32 + 16 + lh * 8];
            }
#pragma unroll
            for (int i = 0; i < 2; ++i)
#pragma unroll
                for (int j = 0; j < 2; ++j)
                    acc[i][j] = wmma_bf16(af[i], bfr[j], acc[i][j]);
        }
        __syncthreads();   // fences this buffer before it is re-filled
    }

    // ---- store (C layout: M = r + 8*lh, N = ln) ----
#pragma unroll
    for (int i = 0; i < 2; ++i)
#pragma unroll
        for (int j = 0; j < 2; ++j)
#pragma unroll
            for (int r = 0; r < 8; ++r) {
                int row = Mt + wm * 32 + i * 16 + lh * 8 + r;
                int col = Nt + wn * 32 + j * 16 + ln;
                float v = acc[i][j][r] * out_scale;
                if (OUT_MODE == 2) {
                    ((float*)Out)[(size_t)row * Ee + col] = v;
                } else if (OUT_MODE == 0) {
                    ((unsigned short*)Out)[(size_t)row * Ee + col] = f2bfu(v);
                } else { // transposed V: [b, h, d, s]
                    int b = row >> 11, s = row & (Ss - 1);
                    int h = col >> 6,  d = col & (Dd - 1);
                    ((unsigned short*)Out)[((size_t)((b * Hh + h) * Dd + d)) * Ss + s] = f2bfu(v);
                }
            }
}

// ============================================================================
// Flash attention. 128 threads = 4 waves; each wave owns a 16-query tile,
// block covers 64 queries of one (b,h). Keys processed 64 at a time:
//   scores  = 8 x WMMA, rowmax = max3 + 4-step shuffle butterfly,
//   P via LDS re-layout (intra-wave DS order), rowsum = 2 x ones-WMMA,
//   O      += 8 x WMMA.
// ============================================================================
__global__ __launch_bounds__(128)
void flash_attn(const unsigned short* __restrict__ qb,
                const unsigned short* __restrict__ kb,
                const unsigned short* __restrict__ vt,
                unsigned short* __restrict__ ao)
{
    __shared__ unsigned short Plds[4][16][64];

    const int tid = threadIdx.x, lane = tid & 31, w = tid >> 5;
    const int ln = lane & 15, lh = lane >> 4;
    const int bh = blockIdx.y;
    const int b = bh >> 4, h = bh & 15;
    const int q0 = blockIdx.x * 64 + w * 16;

    FragU ones;
    {
        const unsigned int o2 = 0x3F803F80u;
        ones.q[0] = uint4{o2, o2, o2, o2};
        ones.q[1] = uint4{o2, o2, o2, o2};
    }

    FragU qf[2];
#pragma unroll
    for (int kk = 0; kk < 2; ++kk) {
        size_t base = (size_t)(b * Ss + q0 + ln) * Ee + h * Dd + kk * 32;
        qf[kk].q[0] = *(const uint4*)&qb[base + lh * 8];
        qf[kk].q[1] = *(const uint4*)&qb[base + 16 + lh * 8];
    }

    float mrow[8], lrow[8], lsc[8];
    v8f oacc[4] = {};
#pragma unroll
    for (int r = 0; r < 8; ++r) { mrow[r] = -1e30f; lrow[r] = 0.0f; }

    for (int kc = 0; kc < Ss; kc += 64) {
        // ---- scores: S[16q x 64k], 4 column subtiles ----
        v8f sacc[4];
        const v8f z = {};
#pragma unroll
        for (int j = 0; j < 4; ++j) {
            FragU kf0, kf1;
            size_t base = (size_t)(b * Ss + kc + j * 16 + ln) * Ee
                        + h * Dd + lh * 8;
            kf0.q[0] = *(const uint4*)&kb[base];
            kf0.q[1] = *(const uint4*)&kb[base + 16];
            kf1.q[0] = *(const uint4*)&kb[base + 32];
            kf1.q[1] = *(const uint4*)&kb[base + 48];
            sacc[j] = wmma_bf16(qf[0], kf0, z);
            sacc[j] = wmma_bf16(qf[1], kf1, sacc[j]);
        }

        // ---- online softmax ----
#pragma unroll
        for (int r = 0; r < 8; ++r) {
            float s0 = sacc[0][r], s1 = sacc[1][r];
            float s2 = sacc[2][r], s3 = sacc[3][r];
            float v = fmaxf(fmaxf(s0, s1), fmaxf(s2, s3));
#pragma unroll
            for (int m = 1; m <= 8; m <<= 1) v = fmaxf(v, __shfl_xor(v, m, 32));
            float mn  = fmaxf(mrow[r], v);
            float scl = __expf(mrow[r] - mn);
            mrow[r] = mn;
            lsc[r]  = scl;
            Plds[w][lh * 8 + r][ln]      = f2bfu(__expf(s0 - mn));
            Plds[w][lh * 8 + r][16 + ln] = f2bfu(__expf(s1 - mn));
            Plds[w][lh * 8 + r][32 + ln] = f2bfu(__expf(s2 - mn));
            Plds[w][lh * 8 + r][48 + ln] = f2bfu(__expf(s3 - mn));
#pragma unroll
            for (int d4 = 0; d4 < 4; ++d4) oacc[d4][r] *= scl;
        }

        // ---- P as A-fragments ----
        FragU pf0, pf1;
        pf0.q[0] = *(const uint4*)&Plds[w][ln][lh * 8];
        pf0.q[1] = *(const uint4*)&Plds[w][ln][16 + lh * 8];
        pf1.q[0] = *(const uint4*)&Plds[w][ln][32 + lh * 8];
        pf1.q[1] = *(const uint4*)&Plds[w][ln][48 + lh * 8];

        // ---- row-sum via ones-WMMA ----
        v8f ssum = wmma_bf16(pf0, ones, z);
        ssum     = wmma_bf16(pf1, ones, ssum);
#pragma unroll
        for (int r = 0; r < 8; ++r) lrow[r] = lrow[r] * lsc[r] + ssum[r];

        // ---- O += P @ V ----
#pragma unroll
        for (int d4 = 0; d4 < 4; ++d4) {
            FragU vf0, vf1;
            size_t base = (size_t)((b * Hh + h) * Dd + d4 * 16 + ln) * Ss
                        + kc + lh * 8;
            vf0.q[0] = *(const uint4*)&vt[base];
            vf0.q[1] = *(const uint4*)&vt[base + 16];
            vf1.q[0] = *(const uint4*)&vt[base + 32];
            vf1.q[1] = *(const uint4*)&vt[base + 48];
            oacc[d4] = wmma_bf16(pf0, vf0, oacc[d4]);
            oacc[d4] = wmma_bf16(pf1, vf1, oacc[d4]);
        }
    }

    // ---- normalize + store (bf16, [B*S, E]) ----
#pragma unroll
    for (int d4 = 0; d4 < 4; ++d4) {
#pragma unroll
        for (int r = 0; r < 8; ++r) {
            float o = oacc[d4][r] / lrow[r];
            ao[(size_t)(b * Ss + q0 + lh * 8 + r) * Ee + h * Dd + d4 * 16 + ln] = f2bfu(o);
        }
    }
}

// ============================================================================
extern "C" void kernel_launch(void* const* d_in, const int* in_sizes, int n_in,
                              void* d_out, int out_size, void* d_ws, size_t ws_size,
                              hipStream_t stream)
{
    const float* query = (const float*)d_in[0];
    const float* key   = (const float*)d_in[1];
    const float* value = (const float*)d_in[2];
    const float* Wq    = (const float*)d_in[3];
    const float* Wk    = (const float*)d_in[4];
    const float* Wv    = (const float*)d_in[5];
    const float* Wo    = (const float*)d_in[6];

    const size_t SZ = (size_t)Mrows * Ee;          // 4M elements
    const size_t WZ = (size_t)Ee * Ee;             // 1M elements
    unsigned short* qb = (unsigned short*)d_ws;    // bf16 Q     (8 MB)
    unsigned short* kb = qb + SZ;                  // bf16 K     (8 MB)
    unsigned short* vt = kb + SZ;                  // bf16 V^T   (8 MB)
    unsigned short* ao = vt + SZ;                  // bf16 attn  (8 MB)
    unsigned short* xb = ao + SZ;                  // bf16 activation staging
    unsigned short* wb = xb + SZ;                  // bf16 weight staging

    dim3 gridP(Ee / 128, Mrows / 64);              // (8, 64)
    dim3 blockP(256);
    const size_t smemP = 2 * (64 * 64 + 128 * 64) * sizeof(unsigned short); // 48 KB
    const int nX8 = (int)(SZ / 8), nW8 = (int)(WZ / 8);

    // Q projection (score scale folded in)
    to_bf16<<<nX8 / 256, 256, 0, stream>>>(query, xb, nX8);
    to_bf16<<<nW8 / 256, 256, 0, stream>>>(Wq, wb, nW8);
    proj_gemm<0><<<gridP, blockP, smemP, stream>>>(xb, wb, qb, INV_SCALE);
    // K projection
    to_bf16<<<nX8 / 256, 256, 0, stream>>>(key, xb, nX8);
    to_bf16<<<nW8 / 256, 256, 0, stream>>>(Wk, wb, nW8);
    proj_gemm<0><<<gridP, blockP, smemP, stream>>>(xb, wb, kb, 1.0f);
    // V projection (stored transposed)
    to_bf16<<<nX8 / 256, 256, 0, stream>>>(value, xb, nX8);
    to_bf16<<<nW8 / 256, 256, 0, stream>>>(Wv, wb, nW8);
    proj_gemm<1><<<gridP, blockP, smemP, stream>>>(xb, wb, vt, 1.0f);

    // attention
    dim3 gridA(Ss / 64, Bb * Hh);                  // (32, 32)
    flash_attn<<<gridA, dim3(128), 0, stream>>>(qb, kb, vt, ao);

    // output projection -> fp32
    to_bf16<<<nW8 / 256, 256, 0, stream>>>(Wo, wb, nW8);
    proj_gemm<2><<<gridP, blockP, smemP, stream>>>(ao, wb, d_out, 1.0f);
}
